// InterpretableMultiHeadAttention_64407329571180
// MI455X (gfx1250) — matrix operand order repaired and checked by reading, not compiled, not run on hardware
//
#include <hip/hip_runtime.h>
#include <hip/hip_bf16.h>

// InterpretableMultiHeadAttention for MI455X (gfx1250, wave32, WMMA).
// - All GEMMs: v_wmma_f32_16x16x32_bf16 (bf16 A/B, f32 accumulate).
// - x and all weights pre-converted (weights pre-transposed) to bf16 once,
//   so every WMMA operand is two aligned global_load_b128 / ds_load_b128.
// - All WMMA inner loops software-pipelined (2 register stages) so loads
//   for step k+1 are in flight while step k's WMMAs execute.
// - Attention probabilities live entirely in LDS (never touch HBM);
//   attn_mean accumulated across heads with LDS f32 atomics.
// - Q stripe per block fetched by the Tensor Data Mover
//   (tensor_load_to_lds + s_wait_tensorcnt) with LDS row padding.
//
// Workspace (~30.6 MB):
//   xb/Obb bf16 [B*T, D]   8 MB   (x bf16; later reused as attn output)
//   Qb     bf16 [B*T, D]   8 MB
//   Kb     bf16 [B*T, D]   8 MB
//   Vt     bf16 [B, DH, T] 0.5 MB (V transposed for P@V B-operand)
//   Wqt/Wkt/Wot bf16 [N,K] 2 MB each, Wvt bf16 [DH,K] 128 KB

#define B_  4
#define T_  1024
#define D_  1024
#define H_  16
#define DH_ 64
#define M_  (B_ * T_)   // 4096

typedef __attribute__((ext_vector_type(16))) __bf16 v16bf;
typedef __attribute__((ext_vector_type(8)))  float  v8f;
typedef __attribute__((ext_vector_type(4)))  unsigned int v4u;
typedef __attribute__((ext_vector_type(8)))  int    v8i;
typedef __attribute__((ext_vector_type(4)))  int    v4i;

union BFV {
    v16bf v;
    uint4 q[2];
    unsigned short u[16];
};

__device__ __forceinline__ unsigned short f2bf(float f) {
    union { float f; unsigned u; } a{f};
    unsigned r = a.u + 0x7FFFu + ((a.u >> 16) & 1u);  // round-to-nearest-even
    return (unsigned short)(r >> 16);
}
__device__ __forceinline__ float bf2f(unsigned short s) {
    union { unsigned u; float f; } a{(unsigned)s << 16};
    return a.f;
}
__device__ __forceinline__ v8f wmma_bf16(const BFV& a, const BFV& b, v8f c) {
    return __builtin_amdgcn_wmma_f32_16x16x32_bf16(
        false, a.v, false, b.v, (short)0, c, false, false);
}

// ---------------------------------------------------------------------------
// Kernel 0a: elementwise f32 -> bf16 (8 elems/thread, b128 in / b128 out)
// ---------------------------------------------------------------------------
__global__ __launch_bounds__(256) void k_cvt(
    const float* __restrict__ src, unsigned short* __restrict__ dst)
{
    const int i = blockIdx.x * 256 + threadIdx.x;   // 8 elements each
    const float4* s = (const float4*)src;
    float4 f0 = s[2 * i], f1 = s[2 * i + 1];
    union { unsigned short u[8]; uint4 q; } p;
    p.u[0] = f2bf(f0.x); p.u[1] = f2bf(f0.y);
    p.u[2] = f2bf(f0.z); p.u[3] = f2bf(f0.w);
    p.u[4] = f2bf(f1.x); p.u[5] = f2bf(f1.y);
    p.u[6] = f2bf(f1.z); p.u[7] = f2bf(f1.w);
    ((uint4*)dst)[i] = p.q;
}

// ---------------------------------------------------------------------------
// Kernel 0b: f32 [K][N] -> bf16 [N][K] (transpose through a 64x64 LDS tile)
// ---------------------------------------------------------------------------
__global__ __launch_bounds__(256) void k_cvt_t(
    const float* __restrict__ src, unsigned short* __restrict__ dst,
    int K, int N)
{
    __shared__ unsigned short tile[64][65];
    const int kb = blockIdx.x * 64, nb = blockIdx.y * 64;
    for (int i = threadIdx.x; i < 64 * 64; i += 256) {
        int r = i >> 6, c = i & 63;               // r: k, c: n (coalesced)
        tile[c][r] = f2bf(src[(size_t)(kb + r) * N + nb + c]);
    }
    __syncthreads();
    for (int i = threadIdx.x; i < 64 * 64; i += 256) {
        int r = i >> 6, c = i & 63;               // r: n, c: k (coalesced)
        dst[(size_t)(nb + r) * K + kb + c] = tile[r][c];
    }
}

// ---------------------------------------------------------------------------
// Kernel 1: fused Q,K projection. 4 waves/block, one 16x16 tile per wave.
// Two-stage software pipeline over the K dimension.
// ---------------------------------------------------------------------------
__global__ __launch_bounds__(128) void k_proj_qk(
    const unsigned short* __restrict__ xb,
    const unsigned short* __restrict__ Wqt, const float* __restrict__ bq,
    const unsigned short* __restrict__ Wkt, const float* __restrict__ bk,
    unsigned short* __restrict__ Qb, unsigned short* __restrict__ Kb)
{
    const int lane = threadIdx.x & 31, wave = threadIdx.x >> 5;
    const int m0 = blockIdx.x * 16;
    const int n0 = blockIdx.y * 64 + wave * 16;
    const int row = lane & 15, g = lane >> 4;

    const unsigned short* ax  = xb  + (size_t)(m0 + row) * D_;  // A: row m
    const unsigned short* bqp = Wqt + (size_t)(n0 + row) * D_;  // B: col n
    const unsigned short* bkp = Wkt + (size_t)(n0 + row) * D_;

    struct Stage { BFV a, bq, bk; };
    auto load_stage = [&](int kt, Stage& s) {
        const int k0 = kt * 32;
        s.a.q[0]  = *(const uint4*)(ax  + k0 + g * 16);
        s.a.q[1]  = *(const uint4*)(ax  + k0 + g * 16 + 8);
        s.bq.q[0] = *(const uint4*)(bqp + k0 + g * 16);
        s.bq.q[1] = *(const uint4*)(bqp + k0 + g * 16 + 8);
        s.bk.q[0] = *(const uint4*)(bkp + k0 + g * 16);
        s.bk.q[1] = *(const uint4*)(bkp + k0 + g * 16 + 8);
    };

    v8f cQ = {}, cK = {};
    Stage cur, nxt;
    load_stage(0, cur);
#pragma unroll 4
    for (int kt = 0; kt < 31; ++kt) {
        load_stage(kt + 1, nxt);                 // in flight over the WMMAs
        __builtin_prefetch(ax + kt * 32 + 256, 0, 1);
        cQ = wmma_bf16(cur.a, cur.bq, cQ);
        cK = wmma_bf16(cur.a, cur.bk, cK);
        cur = nxt;
    }
    cQ = wmma_bf16(cur.a, cur.bq, cQ);
    cK = wmma_bf16(cur.a, cur.bk, cK);

    const int colg = n0 + row;                    // C: col = lane&15
    const float bQ = bq[colg], bK = bk[colg];
#pragma unroll
    for (int r = 0; r < 8; ++r) {                 // C: row = r + 8*g
        const size_t mrow = (size_t)(m0 + r + 8 * g);
        Qb[mrow * D_ + colg] = f2bf(cQ[r] + bQ);
        Kb[mrow * D_ + colg] = f2bf(cK[r] + bK);
    }
}

// ---------------------------------------------------------------------------
// Kernel 2: V projection; output transposed Vt[b][dh][t] (bf16), packed
// 8-wide b128 stores (C rows are consecutive t). Two-stage pipeline.
// ---------------------------------------------------------------------------
__global__ __launch_bounds__(128) void k_proj_v(
    const unsigned short* __restrict__ xb,
    const unsigned short* __restrict__ Wvt, const float* __restrict__ bv,
    unsigned short* __restrict__ Vt)
{
    const int lane = threadIdx.x & 31, wave = threadIdx.x >> 5;
    const int m0 = blockIdx.x * 64 + wave * 16;
    const int n0 = blockIdx.y * 16;
    const int row = lane & 15, g = lane >> 4;

    const unsigned short* ax = xb  + (size_t)(m0 + row) * D_;
    const unsigned short* bp = Wvt + (size_t)(n0 + row) * D_;

    struct Stage { BFV a, b; };
    auto load_stage = [&](int kt, Stage& s) {
        const int k0 = kt * 32;
        s.a.q[0] = *(const uint4*)(ax + k0 + g * 16);
        s.a.q[1] = *(const uint4*)(ax + k0 + g * 16 + 8);
        s.b.q[0] = *(const uint4*)(bp + k0 + g * 16);
        s.b.q[1] = *(const uint4*)(bp + k0 + g * 16 + 8);
    };

    v8f cV = {};
    Stage cur, nxt;
    load_stage(0, cur);
#pragma unroll 4
    for (int kt = 0; kt < 31; ++kt) {
        load_stage(kt + 1, nxt);
        cV = wmma_bf16(cur.a, cur.b, cV);
        cur = nxt;
    }
    cV = wmma_bf16(cur.a, cur.b, cV);

    const int bb = m0 >> 10;
    const int t0 = (m0 & 1023) + 8 * g;
    const float bvv = bv[n0 + row];
    union { unsigned short s[8]; uint4 q; } pk;
#pragma unroll
    for (int r = 0; r < 8; ++r) pk.s[r] = f2bf(cV[r] + bvv);
    *(uint4*)&Vt[((size_t)(bb * DH_ + n0 + row)) * T_ + t0] = pk.q;
}

// ---------------------------------------------------------------------------
// Kernel 3: attention. One block per (b, 16-query stripe); 2 waves x 8 heads.
// Q stripe DMA'd into LDS by the TDM (with 16B/1KB pad -> bank-staggered
// rows). Scores + probs stay in LDS; attn_mean via ds_add_f32 atomics.
// Both WMMA loops two-stage pipelined.
// ---------------------------------------------------------------------------
#define QROW 2080                       // bytes/row: 2048 + 2 pads of 16B
#define SROW 1032
#define PROW 1040
#define OFF_S   (16 * QROW)                         //  33280
#define OFF_P   (OFF_S + 2 * 16 * SROW * 4)         // 165376
#define OFF_ACC (OFF_P + 2 * 16 * PROW * 2)         // 231936
#define OFF_INV (OFF_ACC + 16 * SROW * 4)           // 297984
#define ATTN_SMEM (OFF_INV + 2 * 16 * 4)            // 298112 < 320KB

__global__ __launch_bounds__(64) void k_attn(
    const unsigned short* __restrict__ Qb,
    const unsigned short* __restrict__ Kb,
    const unsigned short* __restrict__ Vt,
    unsigned short* __restrict__ Obb,
    float* __restrict__ attn_mean)
{
    extern __shared__ char smem[];
    const char*     Qs  = smem;                          // [16][QROW bytes]
    float*          S   = (float*)(smem + OFF_S);        // [2][16][SROW]
    unsigned short* P   = (unsigned short*)(smem + OFF_P);
    float*          ACC = (float*)(smem + OFF_ACC);      // [16][SROW]
    float*          INV = (float*)(smem + OFF_INV);      // [2][16]

    const int tid = threadIdx.x, lane = tid & 31, wave = tid >> 5;
    const int q0 = blockIdx.x * 16;
    const int b  = blockIdx.y;
    const int row = lane & 15, g = lane >> 4;

    float*          Sw   = S + wave * 16 * SROW;
    unsigned short* Pw   = P + wave * 16 * PROW;
    float*          INVw = INV + wave * 16;

    // ---- TDM: copy Q stripe [16 x 1024] bf16 -> LDS (padded rows) ----
    if (wave == 0) {
        unsigned long long ga = (unsigned long long)(uintptr_t)Qb +
                                (unsigned long long)(b * T_ + q0) * D_ * 2ull;
        v4u g0;
        g0[0] = 1u;                                   // count=1 (user D#)
        g0[1] = (unsigned)(uintptr_t)Qs;              // lds_addr
        g0[2] = (unsigned)(ga & 0xFFFFFFFFu);         // global_addr lo
        g0[3] = (unsigned)((ga >> 32) & 0x01FFFFFFu) | (2u << 30); // type=2
        v8i g1;
        g1[0] = (int)((1u << 16)       // data_size = 2B
                    | (1u << 20)       // pad_enable
                    | (7u << 22)       // pad_interval: 256 dwords (1KB)
                    | (3u << 25));     // pad_amount: 4 dwords (16B)
        g1[1] = (int)(1024u << 16);    // tensor_dim0 = 1024
        g1[2] = (int)(16u << 16);      // tensor_dim1 = 16
        g1[3] = (int)(1024u << 16);    // tile_dim0 = 1024
        g1[4] = 16;                    // tile_dim1 = 16
        g1[5] = 1024;                  // tensor_dim0_stride
        g1[6] = 0; g1[7] = 0;
        v4i z4 = {0, 0, 0, 0};
        v8i z8 = {0, 0, 0, 0, 0, 0, 0, 0};
        __builtin_amdgcn_tensor_load_to_lds(g0, g1, z4, z4, z8, 0);
        __builtin_amdgcn_s_wait_tensorcnt(0);
    }

    for (int i = tid; i < 16 * SROW; i += 64) ACC[i] = 0.f;
    __syncthreads();

    for (int hh = 0; hh < 8; ++hh) {
        const int h = wave * 8 + hh;
        const unsigned short* Kh = Kb + ((size_t)b * T_) * D_ + h * DH_;

        // Q A-operands from padded LDS: elem e -> byte 2e + 16*(e>>9)
        BFV a0, a1;
        {
            const char* qr = Qs + row * QROW;
            auto ld = [&](int e) -> uint4 {
                return *(const uint4*)(qr + 2 * e + 16 * (e >> 9));
            };
            const int e0 = h * DH_;
            a0.q[0] = ld(e0 + g * 16);
            a0.q[1] = ld(e0 + g * 16 + 8);
            a1.q[0] = ld(e0 + 32 + g * 16);
            a1.q[1] = ld(e0 + 32 + g * 16 + 8);
        }

        // ---- scores: S = (Q Kh^T) * DH^-0.5 (pipelined over key tiles) ----
        auto loadK = [&](int kt, BFV& b0, BFV& b1) {
            const unsigned short* kp = Kh + ((size_t)(kt * 16 + row)) * D_;
            b0.q[0] = *(const uint4*)(kp + g * 16);
            b0.q[1] = *(const uint4*)(kp + g * 16 + 8);
            b1.q[0] = *(const uint4*)(kp + 32 + g * 16);
            b1.q[1] = *(const uint4*)(kp + 32 + g * 16 + 8);
        };
        auto score = [&](int kt, const BFV& b0, const BFV& b1) {
            v8f c = {};
            c = wmma_bf16(a0, b0, c);
            c = wmma_bf16(a1, b1, c);
#pragma unroll
            for (int r = 0; r < 8; ++r)
                Sw[(r + 8 * g) * SROW + kt * 16 + row] = c[r] * 0.125f;
        };
        BFV kb0, kb1, kn0, kn1;
        loadK(0, kb0, kb1);
#pragma unroll 4
        for (int kt = 0; kt < 63; ++kt) {
            loadK(kt + 1, kn0, kn1);
            score(kt, kb0, kb1);
            kb0 = kn0; kb1 = kn1;
        }
        score(63, kb0, kb1);
        __syncthreads();

        // ---- softmax over 1024 keys (lane = one half-row) ----
        const int cb = g * 512;
        float mx = -1e30f;
        for (int c = 0; c < 512; ++c)
            mx = fmaxf(mx, Sw[row * SROW + cb + c]);
        mx = fmaxf(mx, __shfl_xor(mx, 16, 32));
        float sum = 0.f;
        for (int c = 0; c < 512; ++c) {
            float e = __expf(Sw[row * SROW + cb + c] - mx);
            sum += e;
            Pw[row * PROW + cb + c] = f2bf(e);     // unnormalized probs
        }
        sum += __shfl_xor(sum, 16, 32);
        const float inv = 1.f / sum;
        if (g == 0) INVw[row] = inv;
        const float w = inv * (1.f / (float)H_);
        for (int c = 0; c < 512; ++c)
            atomicAdd(&ACC[row * SROW + cb + c],
                      bf2f(Pw[row * PROW + cb + c]) * w);
        __syncthreads();

        // ---- out_h = (P V) * inv (pipelined over key dimension) ----
        const unsigned short* Vb = Vt + (size_t)b * DH_ * T_;
        struct StPV { BFV a, b0, b1, b2, b3; };
        auto loadV = [&](int ks, int nt) -> BFV {
            BFV bm;
            const unsigned short* bp =
                Vb + ((size_t)(nt * 16 + row)) * T_ + ks * 32 + g * 16;
            bm.q[0] = *(const uint4*)bp;
            bm.q[1] = *(const uint4*)(bp + 8);
            return bm;
        };
        auto loadPV = [&](int ks, StPV& s) {
            const unsigned short* ap = &Pw[row * PROW + ks * 32 + g * 16];
            s.a.q[0] = *(const uint4*)ap;
            s.a.q[1] = *(const uint4*)(ap + 8);
            s.b0 = loadV(ks, 0); s.b1 = loadV(ks, 1);
            s.b2 = loadV(ks, 2); s.b3 = loadV(ks, 3);
        };
        v8f o0 = {}, o1 = {}, o2 = {}, o3 = {};
        StPV cur, nxt;
        loadPV(0, cur);
#pragma unroll 4
        for (int ks = 0; ks < 31; ++ks) {
            loadPV(ks + 1, nxt);
            o0 = wmma_bf16(cur.a, cur.b0, o0);
            o1 = wmma_bf16(cur.a, cur.b1, o1);
            o2 = wmma_bf16(cur.a, cur.b2, o2);
            o3 = wmma_bf16(cur.a, cur.b3, o3);
            cur = nxt;
        }
        o0 = wmma_bf16(cur.a, cur.b0, o0);
        o1 = wmma_bf16(cur.a, cur.b1, o1);
        o2 = wmma_bf16(cur.a, cur.b2, o2);
        o3 = wmma_bf16(cur.a, cur.b3, o3);

        float sc[8];
#pragma unroll
        for (int r = 0; r < 8; ++r) sc[r] = INVw[r + 8 * g];
        unsigned short* Op = Obb + ((size_t)(b * T_ + q0)) * D_ + h * DH_;
#pragma unroll
        for (int nt = 0; nt < 4; ++nt) {
            v8f& o = (nt == 0 ? o0 : nt == 1 ? o1 : nt == 2 ? o2 : o3);
#pragma unroll
            for (int r = 0; r < 8; ++r)
                Op[((size_t)(r + 8 * g)) * D_ + nt * 16 + row] =
                    f2bf(o[r] * sc[r]);
        }
        __syncthreads();
    }

    __syncthreads();
    for (int i = tid; i < 16 * 1024; i += 64) {
        int r = i >> 10, c = i & 1023;
        attn_mean[((size_t)(b * T_ + q0 + r)) * T_ + c] = ACC[r * SROW + c];
    }
}

// ---------------------------------------------------------------------------
// Kernel 4: output projection  out = Obb @ Wo + bo  (f32 out to d_out)
// ---------------------------------------------------------------------------
__global__ __launch_bounds__(128) void k_proj_o(
    const unsigned short* __restrict__ Obb,
    const unsigned short* __restrict__ Wot, const float* __restrict__ bo,
    float* __restrict__ out)
{
    const int lane = threadIdx.x & 31, wave = threadIdx.x >> 5;
    const int m0 = blockIdx.x * 16;
    const int n0 = blockIdx.y * 64 + wave * 16;
    const int row = lane & 15, g = lane >> 4;

    const unsigned short* ap = Obb + (size_t)(m0 + row) * D_;
    const unsigned short* bp = Wot + (size_t)(n0 + row) * D_;

    struct Stage { BFV a, b; };
    auto load_stage = [&](int kt, Stage& s) {
        const int k0 = kt * 32;
        s.a.q[0] = *(const uint4*)(ap + k0 + g * 16);
        s.a.q[1] = *(const uint4*)(ap + k0 + g * 16 + 8);
        s.b.q[0] = *(const uint4*)(bp + k0 + g * 16);
        s.b.q[1] = *(const uint4*)(bp + k0 + g * 16 + 8);
    };

    v8f cO = {};
    Stage cur, nxt;
    load_stage(0, cur);
#pragma unroll 4
    for (int kt = 0; kt < 31; ++kt) {
        load_stage(kt + 1, nxt);
        __builtin_prefetch(ap + kt * 32 + 256, 0, 1);
        cO = wmma_bf16(cur.a, cur.b, cO);
        cur = nxt;
    }
    cO = wmma_bf16(cur.a, cur.b, cO);

    const int colg = n0 + row;
    const float bO = bo[colg];
#pragma unroll
    for (int r = 0; r < 8; ++r)
        out[((size_t)(m0 + r + 8 * g)) * D_ + colg] = cO[r] + bO;
}

// ---------------------------------------------------------------------------
extern "C" void kernel_launch(void* const* d_in, const int* in_sizes, int n_in,
                              void* d_out, int out_size, void* d_ws, size_t ws_size,
                              hipStream_t stream) {
    const float* x  = (const float*)d_in[0];
    const float* Wq = (const float*)d_in[1];
    const float* bq = (const float*)d_in[2];
    const float* Wk = (const float*)d_in[3];
    const float* bk = (const float*)d_in[4];
    const float* Wv = (const float*)d_in[5];
    const float* bv = (const float*)d_in[6];
    const float* Wo = (const float*)d_in[7];
    const float* bo = (const float*)d_in[8];

    char* ws = (char*)d_ws;
    unsigned short* xb  = (unsigned short*)ws;                       // 8 MB
    unsigned short* Obb = (unsigned short*)ws;          // aliases xb (dead)
    unsigned short* Qb  = (unsigned short*)(ws + (8u << 20));        // 8 MB
    unsigned short* Kb  = (unsigned short*)(ws + (16u << 20));       // 8 MB
    unsigned short* Vt  = (unsigned short*)(ws + (24u << 20));       // 0.5 MB
    char* w0 = ws + (24u << 20) + (1u << 19);
    unsigned short* Wqt = (unsigned short*)(w0);                     // 2 MB
    unsigned short* Wkt = (unsigned short*)(w0 + (2u << 20));        // 2 MB
    unsigned short* Wot = (unsigned short*)(w0 + (4u << 20));        // 2 MB
    unsigned short* Wvt = (unsigned short*)(w0 + (6u << 20));        // 128 KB

    float* out       = (float*)d_out;                      // [B,T,D]
    float* attn_mean = out + (size_t)B_ * T_ * D_;         // [B,T,T]

    // one-shot precision/layout conversion (all L2-resident afterwards)
    k_cvt<<<(M_ * D_ / 8) / 256, 256, 0, stream>>>(x, xb);
    dim3 gT(D_ / 64, D_ / 64);
    k_cvt_t<<<gT, 256, 0, stream>>>(Wq, Wqt, D_, D_);
    k_cvt_t<<<gT, 256, 0, stream>>>(Wk, Wkt, D_, D_);
    k_cvt_t<<<gT, 256, 0, stream>>>(Wo, Wot, D_, D_);
    dim3 gTv(D_ / 64, DH_ / 64);
    k_cvt_t<<<gTv, 256, 0, stream>>>(Wv, Wvt, D_, DH_);

    dim3 gQK(M_ / 16, D_ / 64);            // 256 x 16
    k_proj_qk<<<gQK, 128, 0, stream>>>(xb, Wqt, bq, Wkt, bk, Qb, Kb);

    dim3 gV(M_ / 64, DH_ / 16);            // 64 x 4
    k_proj_v<<<gV, 128, 0, stream>>>(xb, Wvt, bv, Vt);

    (void)hipFuncSetAttribute(reinterpret_cast<const void*>(&k_attn),
                              hipFuncAttributeMaxDynamicSharedMemorySize,
                              ATTN_SMEM);
    dim3 gA(T_ / 16, B_);                  // 64 x 4
    k_attn<<<gA, 64, ATTN_SMEM, stream>>>(Qb, Kb, Vt, Obb, attn_mean);

    dim3 gO(M_ / 16, D_ / 64);             // 256 x 16
    k_proj_o<<<gO, 128, 0, stream>>>(Obb, Wot, bo, out);
}
